// TextGuidedGAT_38895223832808
// MI455X (gfx1250) — matrix operand order repaired and checked by reading, not compiled, not run on hardware
//
#include <hip/hip_runtime.h>
#include <hip/hip_bf16.h>

// Problem constants (match reference)
#define B_   2
#define C_   256
#define N_   4096
#define DT_  512
#define OUT_ 256
#define H_   4
#define HD_  64
#define MT_  64   // keys per attention iteration

typedef __attribute__((ext_vector_type(16))) _Float16 v16h;
typedef __attribute__((ext_vector_type(8)))  _Float16 v8h;
typedef __attribute__((ext_vector_type(8)))  float    v8f;

// Probe for CDNA5 async global->LDS DMA builtin (guarded: falls back cleanly)
#if defined(__has_builtin)
#if __has_builtin(__builtin_amdgcn_global_load_async_to_lds_b128)
#define HAVE_ASYNC_LDS 1
#endif
#endif
#ifndef HAVE_ASYNC_LDS
#define HAVE_ASYNC_LDS 0
#endif

#if HAVE_ASYNC_LDS
// Signature (from clang diagnostics): param0 = AS1 int4*, param1 = AS3 int4*,
// then i32 offset, i32 cpol.
typedef int vi4_t __attribute__((ext_vector_type(4)));
typedef __attribute__((address_space(1))) vi4_t* gas_ptr;
typedef __attribute__((address_space(3))) vi4_t* las_ptr;
#endif

static __device__ __forceinline__ v16h cat16(v8h lo, v8h hi) {
  v16h r;
#pragma unroll
  for (int i = 0; i < 8; ++i) { r[i] = lo[i]; r[i + 8] = hi[i]; }
  return r;
}

// ---- 16-lane butterfly reductions in pure VALU via DPP -------------------
static __device__ __forceinline__ float red_max16(float x) {
  x = fmaxf(x, __int_as_float(__builtin_amdgcn_mov_dpp(__float_as_int(x), 0xB1, 0xF, 0xF, true)));
  x = fmaxf(x, __int_as_float(__builtin_amdgcn_mov_dpp(__float_as_int(x), 0x4E, 0xF, 0xF, true)));
  x = fmaxf(x, __int_as_float(__builtin_amdgcn_mov_dpp(__float_as_int(x), 0x141, 0xF, 0xF, true)));
  x = fmaxf(x, __int_as_float(__builtin_amdgcn_mov_dpp(__float_as_int(x), 0x140, 0xF, 0xF, true)));
  return x;
}
static __device__ __forceinline__ float red_add16(float x) {
  x += __int_as_float(__builtin_amdgcn_mov_dpp(__float_as_int(x), 0xB1, 0xF, 0xF, true));
  x += __int_as_float(__builtin_amdgcn_mov_dpp(__float_as_int(x), 0x4E, 0xF, 0xF, true));
  x += __int_as_float(__builtin_amdgcn_mov_dpp(__float_as_int(x), 0x141, 0xF, 0xF, true));
  x += __int_as_float(__builtin_amdgcn_mov_dpp(__float_as_int(x), 0x140, 0xF, 0xF, true));
  return x;
}

// ---------------------------------------------------------------------------
// 1) t = GELU(text @ Wt^T + bt)   [B, C]  (exact erf GELU)
// ---------------------------------------------------------------------------
__global__ void k_textproj(const float* __restrict__ text,
                           const float* __restrict__ Wt,
                           const float* __restrict__ bt,
                           float* __restrict__ t) {
  const int b = blockIdx.x;
  const int o = threadIdx.x;            // 256 threads = C_
  const float* xr = text + (size_t)b * DT_;
  const float* wr = Wt + (size_t)o * DT_;
  float acc = bt[o];
  for (int c = 0; c < DT_; ++c) acc += xr[c] * wr[c];
  t[b * C_ + o] = 0.5f * acc * (1.0f + erff(acc * 0.70710678118654752f));
}

// ---------------------------------------------------------------------------
// 2) fp32 -> fp16 weight conversion
// ---------------------------------------------------------------------------
__global__ void k_cvt(const float* __restrict__ src, _Float16* __restrict__ dst, int n) {
  int i = blockIdx.x * blockDim.x + threadIdx.x;
  if (i < n) dst[i] = (_Float16)src[i];
}

// ---------------------------------------------------------------------------
// 3) LDS-tiled transpose: pf [B,C,N] f32 -> pfT [B,N,C] f16 and
//    fusedT [B,N,C] f16 (fused = pf + t[b,c]). Coalesced both directions.
// ---------------------------------------------------------------------------
__global__ void k_transpose_fuse(const float* __restrict__ pf, const float* __restrict__ t,
                                 _Float16* __restrict__ pfT, _Float16* __restrict__ fT) {
  __shared__ float tile[64][65];
  const int n0 = blockIdx.x * 64, c0 = blockIdx.y * 64, b = blockIdx.z;
  const int tid = threadIdx.x;

  {
    const int c = tid >> 2, ns = (tid & 3) * 16;
    const float* src = pf + ((size_t)b * C_ + c0 + c) * N_ + n0 + ns;
#pragma unroll
    for (int i = 0; i < 4; ++i)
      *(float4*)&tile[c][ns + i * 4] = *(const float4*)(src + i * 4);
  }
  __syncthreads();
  {
    const int n = tid >> 2, cs = (tid & 3) * 16;
    const float* tb = t + (size_t)b * C_;
    alignas(16) _Float16 outp[16], outf[16];
#pragma unroll
    for (int i = 0; i < 16; ++i) {
      float v = tile[cs + i][n];
      outp[i] = (_Float16)v;
      outf[i] = (_Float16)(v + tb[c0 + cs + i]);
    }
    _Float16* dp = pfT + ((size_t)b * N_ + n0 + n) * C_ + c0 + cs;
    _Float16* df = fT  + ((size_t)b * N_ + n0 + n) * C_ + c0 + cs;
#pragma unroll
    for (int i = 0; i < 2; ++i) {
      *(float4*)(dp + i * 8) = *(float4*)&outp[i * 8];
      *(float4*)(df + i * 8) = *(float4*)&outf[i * 8];
    }
  }
}

// ---------------------------------------------------------------------------
// 4) Generic WMMA GEMM: Y = act(A[tok,256] @ W^T + bias), f16 output.
//    OUTMODE 0: token-major Y[tok*256 + o] ; OUTMODE 1: o-major (for V)
// ---------------------------------------------------------------------------
template <int OUTMODE, bool RELU>
__launch_bounds__(128, 1)
__global__ void k_gemm(const _Float16* __restrict__ A, const _Float16* __restrict__ Wh,
                       const float* __restrict__ bias, _Float16* __restrict__ Y) {
  const int tid = threadIdx.x, wave = tid >> 5, lane = tid & 31;
  const int lh = lane >> 4, l16 = lane & 15;
  const int tok0 = (blockIdx.x * 4 + wave) * 16;
  const int o0   = blockIdx.y * 64;

  v8f acc[4];
#pragma unroll
  for (int dt = 0; dt < 4; ++dt)
#pragma unroll
    for (int r = 0; r < 8; ++r) acc[dt][r] = 0.f;

  for (int kc = 0; kc < C_; kc += 32) {
    const _Float16* xr = A + (size_t)(tok0 + l16) * C_ + kc + lh * 8;
    v16h aX = cat16(*(const v8h*)xr, *(const v8h*)(xr + 16));
#pragma unroll
    for (int dt = 0; dt < 4; ++dt) {
      const _Float16* wr = Wh + (size_t)(o0 + dt * 16 + l16) * C_ + kc + lh * 8;
      v16h bW = cat16(*(const v8h*)wr, *(const v8h*)(wr + 16));
      acc[dt] = __builtin_amdgcn_wmma_f32_16x16x32_f16(false, aX, false, bW,
                                                       (short)0, acc[dt], false, false);
    }
  }
#pragma unroll
  for (int dt = 0; dt < 4; ++dt) {
    const int o = o0 + dt * 16 + l16;
    const float bi = bias[o];
    if (OUTMODE == 0) {
#pragma unroll
      for (int r = 0; r < 8; ++r) {
        float v = acc[dt][r] + bi;
        if (RELU) v = fmaxf(v, 0.f);
        Y[(size_t)(tok0 + r + lh * 8) * OUT_ + o] = (_Float16)v;
      }
    } else {
      alignas(16) _Float16 pk[8];
      const int toks = tok0 + lh * 8;
      const int b = toks >> 12, n = toks & (N_ - 1);
#pragma unroll
      for (int r = 0; r < 8; ++r) {
        float v = acc[dt][r] + bi;
        if (RELU) v = fmaxf(v, 0.f);
        pk[r] = (_Float16)v;
      }
      *(float4*)(Y + ((size_t)b * OUT_ + o) * N_ + n) = *(float4*)pk;
    }
  }
}

// ---------------------------------------------------------------------------
// 5) Flash attention. 128 threads (4 waves), 64 query rows/WG, 64 keys/iter.
//    K/V tiles double-buffered in LDS, staged by the CDNA5 async global->LDS
//    DMA (ASYNCcnt, zero VGPR cost); fallback: prefetch + late load/store.
// ---------------------------------------------------------------------------
__launch_bounds__(128, 1)
__global__ void k_attn(const _Float16* __restrict__ Qh, const _Float16* __restrict__ Kh,
                       const _Float16* __restrict__ Vh, _Float16* __restrict__ Oh) {
  __shared__ _Float16 Kt[2][64 * 72];    // [m][d], row stride 72 halfs
  __shared__ _Float16 Vt[2][64 * 72];    // [d][m], row stride 72 halfs
  __shared__ _Float16 Pb[4][16 * 72];    // per-wave P [n][m]

  const int tid  = threadIdx.x;
  const int wave = tid >> 5;
  const int lane = tid & 31;
  const int lh   = lane >> 4;
  const int l16  = lane & 15;
  const int b    = blockIdx.z;
  const int hh   = blockIdx.y;
  const int n0   = blockIdx.x * 64 + wave * 16;

  const _Float16* Qbase = Qh + (size_t)b * N_ * OUT_ + hh * HD_;
  const _Float16* Kbase = Kh + (size_t)b * N_ * OUT_ + hh * HD_;
  const _Float16* Vbase = Vh + ((size_t)b * OUT_ + hh * HD_) * N_;

  // Q A-fragments: row n = n0+l16, kbase = lh*8, two 32-d chunks
  v16h aQ[2];
  {
    const _Float16* qrow = Qbase + (size_t)(n0 + l16) * OUT_ + lh * 8;
#pragma unroll
    for (int dc = 0; dc < 2; ++dc)
      aQ[dc] = cat16(*(const v8h*)(qrow + dc * 32), *(const v8h*)(qrow + dc * 32 + 16));
  }

  v8f oacc[4];
#pragma unroll
  for (int dt = 0; dt < 4; ++dt)
#pragma unroll
    for (int r = 0; r < 8; ++r) oacc[dt][r] = 0.f;
  float mrow[8], lrow[8];
#pragma unroll
  for (int r = 0; r < 8; ++r) { mrow[r] = -1e30f; lrow[r] = 0.f; }

  // Staging geometry: thread t covers row (t>>1), 32 halfs at (t&1)*32.
  const int srow = tid >> 1, sseg = (tid & 1) * 32;

#if HAVE_ASYNC_LDS
  auto asyncTile = [&](int m0, int buf) {
    const _Float16* ks = Kbase + (size_t)(m0 + srow) * OUT_ + sseg;
    const _Float16* vs = Vbase + (size_t)srow * N_ + m0 + sseg;
    _Float16* kd = &Kt[buf][srow * 72 + sseg];
    _Float16* vd = &Vt[buf][srow * 72 + sseg];
#pragma unroll
    for (int i = 0; i < 4; ++i) {
      __builtin_amdgcn_global_load_async_to_lds_b128(
          (gas_ptr)(uintptr_t)(ks + i * 8),
          (las_ptr)(unsigned)(uintptr_t)(kd + i * 8),
          0, 0);
      __builtin_amdgcn_global_load_async_to_lds_b128(
          (gas_ptr)(uintptr_t)(vs + i * 8),
          (las_ptr)(unsigned)(uintptr_t)(vd + i * 8),
          0, 0);
    }
  };
  asyncTile(0, 0);
  asm volatile("s_wait_asynccnt 0" ::: "memory");
  __syncthreads();
#else
  auto copyTile = [&](int m0, int buf) {
    const _Float16* ks = Kbase + (size_t)(m0 + srow) * OUT_ + sseg;
    const _Float16* vs = Vbase + (size_t)srow * N_ + m0 + sseg;
    _Float16* kd = &Kt[buf][srow * 72 + sseg];
    _Float16* vd = &Vt[buf][srow * 72 + sseg];
    float4 kr[4], vr[4];
#pragma unroll
    for (int i = 0; i < 4; ++i) kr[i] = *(const float4*)(ks + i * 8);
#pragma unroll
    for (int i = 0; i < 4; ++i) vr[i] = *(const float4*)(vs + i * 8);
#pragma unroll
    for (int i = 0; i < 4; ++i) *(float4*)(kd + i * 8) = kr[i];
#pragma unroll
    for (int i = 0; i < 4; ++i) *(float4*)(vd + i * 8) = vr[i];
  };
  copyTile(0, 0);
  __syncthreads();
#endif

  const int NIT = N_ / MT_;
  for (int it = 0; it < NIT; ++it) {
    const int cur = it & 1;
#if HAVE_ASYNC_LDS
    if (it + 1 < NIT) asyncTile((it + 1) * MT_, 1 - cur);  // DMA, no VGPRs
#else
    if (it + 1 < NIT) {  // warm caches for the post-compute copy
      const int m0n = (it + 1) * MT_;
      __builtin_prefetch(Kbase + (size_t)(m0n + srow) * OUT_ + sseg, 0, 0);
      __builtin_prefetch(Vbase + (size_t)srow * N_ + m0n + sseg, 0, 0);
    }
#endif

    // S = Q^T K : 4 m-sub-tiles x 2 K-chunks = 8 WMMA
    v8f s[4];
#pragma unroll
    for (int msub = 0; msub < 4; ++msub) {
      v8f c = {};
#pragma unroll
      for (int dc = 0; dc < 2; ++dc) {
        const _Float16* p = &Kt[cur][(msub * 16 + l16) * 72 + dc * 32 + lh * 8];
        v16h bfr = cat16(*(const v8h*)p, *(const v8h*)(p + 16));
        c = __builtin_amdgcn_wmma_f32_16x16x32_f16(false, aQ[dc], false, bfr,
                                                   (short)0, c, false, false);
      }
      s[msub] = c;
    }

    // Online softmax per row (DPP reductions); fold rescale into the loop
#pragma unroll
    for (int r = 0; r < 8; ++r) {
      float mx = fmaxf(fmaxf(s[0][r], s[1][r]), fmaxf(s[2][r], s[3][r]));
      mx = red_max16(mx);
      const float mnew = fmaxf(mrow[r], mx);
      const float a = __expf(mrow[r] - mnew);
      float rs = 0.f;
#pragma unroll
      for (int msub = 0; msub < 4; ++msub) {
        float p = __expf(s[msub][r] - mnew);
        s[msub][r] = p;
        rs += p;
      }
      rs = red_add16(rs);
      lrow[r] = lrow[r] * a + rs;
      mrow[r] = mnew;
#pragma unroll
      for (int dt = 0; dt < 4; ++dt) oacc[dt][r] *= a;
    }

    // Transpose P (C-layout -> A-layout) through per-wave LDS
    _Float16* pb = &Pb[wave][0];
#pragma unroll
    for (int msub = 0; msub < 4; ++msub)
#pragma unroll
      for (int r = 0; r < 8; ++r)
        pb[(r + lh * 8) * 72 + msub * 16 + l16] = (_Float16)s[msub][r];

    v16h aP[2];
#pragma unroll
    for (int kc = 0; kc < 2; ++kc) {
      const _Float16* p = pb + l16 * 72 + kc * 32 + lh * 8;
      aP[kc] = cat16(*(const v8h*)p, *(const v8h*)(p + 16));
    }

    // O += P @ V : 4 d-tiles x 2 K-chunks = 8 WMMA
#pragma unroll
    for (int dt = 0; dt < 4; ++dt)
#pragma unroll
      for (int kc = 0; kc < 2; ++kc) {
        const _Float16* p = &Vt[cur][(dt * 16 + l16) * 72 + kc * 32 + lh * 8];
        v16h bV = cat16(*(const v8h*)p, *(const v8h*)(p + 16));
        oacc[dt] = __builtin_amdgcn_wmma_f32_16x16x32_f16(false, aP[kc], false, bV,
                                                          (short)0, oacc[dt], false, false);
      }

#if HAVE_ASYNC_LDS
    asm volatile("s_wait_asynccnt 0" ::: "memory");
#else
    if (it + 1 < NIT) copyTile((it + 1) * MT_, 1 - cur);  // short register lifetime
#endif
    __syncthreads();
  }

  // Normalize, store token-major [B,N,256] f16
  float inv[8];
#pragma unroll
  for (int r = 0; r < 8; ++r) inv[r] = 1.0f / lrow[r];
  _Float16* obase = Oh + (size_t)b * N_ * OUT_;
#pragma unroll
  for (int dt = 0; dt < 4; ++dt) {
    const int o = hh * HD_ + dt * 16 + l16;
#pragma unroll
    for (int r = 0; r < 8; ++r)
      obase[(size_t)(n0 + r + lh * 8) * OUT_ + o] = (_Float16)(oacc[dt][r] * inv[r]);
  }
}

// ---------------------------------------------------------------------------
// 6) MLP layer 2 + gamma * (.) + residual (f32 out, vectorized epilogue)
// ---------------------------------------------------------------------------
__launch_bounds__(128, 1)
__global__ void k_mlp2(const _Float16* __restrict__ Hh, const _Float16* __restrict__ W2h,
                       const float* __restrict__ b2, const float* __restrict__ gamma,
                       const float* __restrict__ pf, float* __restrict__ out) {
  const int tid = threadIdx.x, wave = tid >> 5, lane = tid & 31;
  const int lh = lane >> 4, l16 = lane & 15;
  const int tok0 = (blockIdx.x * 4 + wave) * 16;
  const int o0   = blockIdx.y * 64;

  v8f acc[4];
#pragma unroll
  for (int dt = 0; dt < 4; ++dt)
#pragma unroll
    for (int r = 0; r < 8; ++r) acc[dt][r] = 0.f;

  for (int kc = 0; kc < OUT_; kc += 32) {
    const _Float16* xr = Hh + (size_t)(tok0 + l16) * OUT_ + kc + lh * 8;
    v16h aX = cat16(*(const v8h*)xr, *(const v8h*)(xr + 16));
#pragma unroll
    for (int dt = 0; dt < 4; ++dt) {
      const _Float16* wr = W2h + (size_t)(o0 + dt * 16 + l16) * OUT_ + kc + lh * 8;
      v16h bW = cat16(*(const v8h*)wr, *(const v8h*)(wr + 16));
      acc[dt] = __builtin_amdgcn_wmma_f32_16x16x32_f16(false, aX, false, bW,
                                                       (short)0, acc[dt], false, false);
    }
  }
  const float g = gamma[0];
  const int toks = tok0 + lh * 8;
  const int b = toks >> 12, n = toks & (N_ - 1);
#pragma unroll
  for (int dt = 0; dt < 4; ++dt) {
    const int o = o0 + dt * 16 + l16;
    const float bi = b2[o];
    const size_t base = ((size_t)b * OUT_ + o) * N_ + n;
    float4 p0 = *(const float4*)(pf + base);
    float4 p1 = *(const float4*)(pf + base + 4);
    float4 y0, y1;
    y0.x = g * (acc[dt][0] + bi) + p0.x;
    y0.y = g * (acc[dt][1] + bi) + p0.y;
    y0.z = g * (acc[dt][2] + bi) + p0.z;
    y0.w = g * (acc[dt][3] + bi) + p0.w;
    y1.x = g * (acc[dt][4] + bi) + p1.x;
    y1.y = g * (acc[dt][5] + bi) + p1.y;
    y1.z = g * (acc[dt][6] + bi) + p1.z;
    y1.w = g * (acc[dt][7] + bi) + p1.w;
    *(float4*)(out + base)     = y0;
    *(float4*)(out + base + 4) = y1;
  }
}

// ---------------------------------------------------------------------------
// Host-side launcher
// ---------------------------------------------------------------------------
extern "C" void kernel_launch(void* const* d_in, const int* in_sizes, int n_in,
                              void* d_out, int out_size, void* d_ws, size_t ws_size,
                              hipStream_t stream) {
  (void)in_sizes; (void)n_in; (void)out_size; (void)ws_size;
  const float* pf    = (const float*)d_in[0];
  const float* text  = (const float*)d_in[1];
  const float* Wt    = (const float*)d_in[2];
  const float* bt    = (const float*)d_in[3];
  const float* Wq    = (const float*)d_in[4];
  const float* bq    = (const float*)d_in[5];
  const float* Wk    = (const float*)d_in[6];
  const float* bk    = (const float*)d_in[7];
  const float* Wv    = (const float*)d_in[8];
  const float* bv    = (const float*)d_in[9];
  const float* W1    = (const float*)d_in[10];
  const float* b1    = (const float*)d_in[11];
  const float* W2    = (const float*)d_in[12];
  const float* b2    = (const float*)d_in[13];
  const float* gamma = (const float*)d_in[14];
  float* out = (float*)d_out;

  char* ws = (char*)d_ws;
  size_t off = 0;
  auto take = [&](size_t bytes) -> void* {
    void* p = ws + off;
    off += (bytes + 255) & ~(size_t)255;
    return p;
  };
  float*    t_  = (float*)   take((size_t)B_ * C_ * sizeof(float));
  _Float16* pfT = (_Float16*)take((size_t)B_ * N_ * C_ * sizeof(_Float16));
  _Float16* fT  = (_Float16*)take((size_t)B_ * N_ * C_ * sizeof(_Float16));
  _Float16* Qh  = (_Float16*)take((size_t)B_ * N_ * OUT_ * sizeof(_Float16));
  _Float16* Kh  = (_Float16*)take((size_t)B_ * N_ * OUT_ * sizeof(_Float16));
  _Float16* Vh  = (_Float16*)take((size_t)B_ * OUT_ * N_ * sizeof(_Float16));
  _Float16* Xh  = (_Float16*)take((size_t)B_ * N_ * OUT_ * sizeof(_Float16));
  _Float16* Hh  = (_Float16*)take((size_t)B_ * N_ * OUT_ * sizeof(_Float16));
  _Float16* Wqh = (_Float16*)take((size_t)OUT_ * C_ * sizeof(_Float16));
  _Float16* Wkh = (_Float16*)take((size_t)OUT_ * C_ * sizeof(_Float16));
  _Float16* Wvh = (_Float16*)take((size_t)OUT_ * C_ * sizeof(_Float16));
  _Float16* W1h = (_Float16*)take((size_t)OUT_ * OUT_ * sizeof(_Float16));
  _Float16* W2h = (_Float16*)take((size_t)OUT_ * OUT_ * sizeof(_Float16));

  k_textproj<<<B_, C_, 0, stream>>>(text, Wt, bt, t_);

  k_cvt<<<(OUT_ * C_ + 255) / 256, 256, 0, stream>>>(Wq, Wqh, OUT_ * C_);
  k_cvt<<<(OUT_ * C_ + 255) / 256, 256, 0, stream>>>(Wk, Wkh, OUT_ * C_);
  k_cvt<<<(OUT_ * C_ + 255) / 256, 256, 0, stream>>>(Wv, Wvh, OUT_ * C_);
  k_cvt<<<(OUT_ * OUT_ + 255) / 256, 256, 0, stream>>>(W1, W1h, OUT_ * OUT_);
  k_cvt<<<(OUT_ * OUT_ + 255) / 256, 256, 0, stream>>>(W2, W2h, OUT_ * OUT_);

  k_transpose_fuse<<<dim3(N_ / 64, C_ / 64, B_), 256, 0, stream>>>(pf, t_, pfT, fT);

  const dim3 ggrid((B_ * N_) / 64, OUT_ / 64);
  k_gemm<0, false><<<ggrid, 128, 0, stream>>>(pfT, Wqh, bq, Qh);
  k_gemm<0, false><<<ggrid, 128, 0, stream>>>(fT,  Wkh, bk, Kh);
  k_gemm<1, false><<<ggrid, 128, 0, stream>>>(pfT, Wvh, bv, Vh);

  k_attn<<<dim3(N_ / 64, H_, B_), 128, 0, stream>>>(Qh, Kh, Vh, Xh);

  k_gemm<0, true><<<ggrid, 128, 0, stream>>>(Xh, W1h, b1, Hh);

  k_mlp2<<<ggrid, 128, 0, stream>>>(Hh, W2h, b2, gamma, pf, out);
}
